// OVFNet_63608465654579
// MI455X (gfx1250) — matrix operand order repaired and checked by reading, not compiled, not run on hardware
//
#include <hip/hip_runtime.h>

// Problem constants: B=64, N=197, C=768, R=16
#define M_ROWS 12608   // B*N
#define K_DIM  768     // C
#define N_OUT  2304    // 3C

typedef unsigned int       u32;
typedef unsigned short     u16;
typedef unsigned long long u64;
typedef __attribute__((ext_vector_type(4)))  u32    u32x4;
typedef __attribute__((ext_vector_type(2)))  u32    u32x2;
typedef __attribute__((ext_vector_type(4)))  int    i32x4;
typedef __attribute__((ext_vector_type(8)))  int    i32x8;
typedef __attribute__((ext_vector_type(4)))  float  f32x4;
typedef __attribute__((ext_vector_type(16))) __bf16 v16bf;
typedef __attribute__((ext_vector_type(8)))  float  v8f;

__device__ __forceinline__ u32 f2bf_rne(float f) {
  u32 u = __float_as_uint(f);
  return (u + 0x7FFFu + ((u >> 16) & 1u)) >> 16;  // round-to-nearest-even bf16
}
__device__ __forceinline__ float bf2f(u32 h) { return __uint_as_float(h << 16); }

// ---------------------------------------------------------------------------
// Kernel 1: fold LoRA into the dense weight, store bf16 hi/lo split planes.
// W_eff[o, c] = w_qkv[o, c] + sum_r Bsel[o', r] * Asel[r, c]
// ---------------------------------------------------------------------------
__global__ void lora_fold_split(const float* __restrict__ w,
                                const float* __restrict__ aq, const float* __restrict__ bq,
                                const float* __restrict__ ak, const float* __restrict__ bk,
                                const float* __restrict__ av, const float* __restrict__ bv,
                                u16* __restrict__ whi, u16* __restrict__ wlo) {
  const int c = blockIdx.x * 256 + threadIdx.x;  // 0..767
  const int o = blockIdx.y;                      // 0..2303
  const float* A; const float* Bm; int oo;
  if (o < 768)        { A = aq; Bm = bq; oo = o; }
  else if (o < 1536)  { A = ak; Bm = bk; oo = o - 768; }
  else                { A = av; Bm = bv; oo = o - 1536; }

  float acc = w[(size_t)o * K_DIM + c];
  #pragma unroll
  for (int r = 0; r < 16; ++r)
    acc = fmaf(Bm[oo * 16 + r], A[r * K_DIM + c], acc);

  const u32 hi = f2bf_rne(acc);
  const u32 lo = f2bf_rne(acc - bf2f(hi));
  whi[(size_t)o * K_DIM + c] = (u16)hi;
  wlo[(size_t)o * K_DIM + c] = (u16)lo;
}

// ---------------------------------------------------------------------------
// TDM: one descriptor moves a 128x32 bf16 W tile (128 rows x 64B, row stride
// 1536B) global -> LDS. Issued by one wave; tracked by TENSORcnt.
// D# layout per CDNA5 ISA ch.8 (group0/group1; groups 2-3 zero for 2D tile).
// ---------------------------------------------------------------------------
__device__ __forceinline__ void tdm_load_w_tile(const u16* __restrict__ wplane,
                                                u16* ldsdst, int nBase, int k0) {
  const u64 g   = (u64)(uintptr_t)(wplane + (size_t)nBase * K_DIM + k0);
  const u32 lds = (u32)(uintptr_t)ldsdst;

  u32x4 g0;
  g0.x = 1u;                                 // count=1, user descriptor
  g0.y = lds;                                // lds_addr (bytes)
  g0.z = (u32)g;                             // global_addr[31:0]
  g0.w = (u32)(g >> 32) | 0x80000000u;       // global_addr[56:32] | type=2

  i32x8 g1;
  g1[0] = (int)(2u << 16);                   // data_size=2 (4-byte units)
  g1[1] = (int)((u32)(K_DIM / 2) << 16);     // tensor_dim0 = 384 units (lo16)
  g1[2] = (int)((u32)N_OUT << 16);           // tensor_dim1 = 2304 rows (lo16)
  g1[3] = (int)(16u << 16);                  // tile_dim0 = 16 units (64B)
  g1[4] = 128;                               // tile_dim1 = 128 rows
  g1[5] = K_DIM / 2;                         // tensor_dim0_stride = 384 units
  g1[6] = 0;
  g1[7] = 0;

  const i32x4 gz = {0, 0, 0, 0};
#if __clang_major__ >= 23
  const i32x8 gz8 = {0, 0, 0, 0, 0, 0, 0, 0};
  __builtin_amdgcn_tensor_load_to_lds(g0, g1, gz, gz, gz8, 0);
#else
  __builtin_amdgcn_tensor_load_to_lds(g0, g1, gz, gz, 0);
#endif
}

// Async global->LDS copy of one 128x32 fp32 X tile (ASYNCcnt-tracked).
__device__ __forceinline__ void async_copy_x_tile(const float* __restrict__ x,
                                                  float* ldsbuf, int mBase, int k0, int tid) {
  #pragma unroll
  for (int i = 0; i < 4; ++i) {
    const int f   = tid + i * 256;     // float4 id 0..1023
    const int row = f >> 3;            // 0..127
    const int c4  = (f & 7) << 2;      // 0,4,..,28
    int gr = mBase + row; if (gr > M_ROWS - 1) gr = M_ROWS - 1;
    const u32 ldsOff = (u32)(uintptr_t)(&ldsbuf[row * 32 + c4]);
    const u64 gaddr  = (u64)(uintptr_t)(x + (size_t)gr * K_DIM + k0 + c4);
    asm volatile("global_load_async_to_lds_b128 %0, %1, off"
                 :: "v"(ldsOff), "v"(gaddr) : "memory");
  }
}

// LDS fp32 tile -> bf16 hi/lo split planes in LDS.
__device__ __forceinline__ void convert_x_tile(const float* ldsraw,
                                               u16* ahi, u16* alo, int tid) {
  #pragma unroll
  for (int i = 0; i < 4; ++i) {
    const int f   = tid + i * 256;
    const int row = f >> 3;
    const int c4  = (f & 7) << 2;
    const f32x4 xv = *(const f32x4*)(&ldsraw[row * 32 + c4]);
    u32 h[4], l[4];
    #pragma unroll
    for (int j = 0; j < 4; ++j) {
      h[j] = f2bf_rne(xv[j]);
      l[j] = f2bf_rne(xv[j] - bf2f(h[j]));
    }
    const u32x2 hp = {h[0] | (h[1] << 16), h[2] | (h[3] << 16)};
    const u32x2 lp = {l[0] | (l[1] << 16), l[2] | (l[3] << 16)};
    *(u32x2*)(&ahi[row * 32 + c4]) = hp;
    *(u32x2*)(&alo[row * 32 + c4]) = lp;
  }
}

// ---------------------------------------------------------------------------
// Kernel 2: out[M, 3C] = X @ W_effT + bias, bf16 hi/lo split (3 WMMAs/tile).
// Block tile 128x128, K-step 32, 8 wave32s; wave tile 32x64 (2x4 WMMA tiles).
// 3-deep pipeline: W via TDM (wave 0), X raw via async-to-LDS two steps
// ahead, bf16-split conversion overlapped behind the WMMAs.
// ---------------------------------------------------------------------------
__global__ void __launch_bounds__(256)
qkv_lora_gemm(const float* __restrict__ x,
              const u16* __restrict__ whi, const u16* __restrict__ wlo,
              const float* __restrict__ bias, float* __restrict__ out) {
  // A: [row 0..127][k 0..31], B: [n 0..127][k 0..31] (== W layout): every WMMA
  // fragment is two contiguous 16B ds reads per lane. 2 pipeline buffers each.
  __shared__ __align__(16) u16   Ahi[2][128 * 32];
  __shared__ __align__(16) u16   Alo[2][128 * 32];
  __shared__ __align__(16) u16   Bhi[2][128 * 32];
  __shared__ __align__(16) u16   Blo[2][128 * 32];
  __shared__ __align__(16) float Xraw[2][128 * 32];

  const int tid    = threadIdx.x;
  const int lane   = tid & 31;
  const int wid    = tid >> 5;
  const int wave_m = wid & 3;    // 4 waves along M (4*32 = 128)
  const int wave_n = wid >> 2;   // 2 waves along N (2*64 = 128)
  const int lhalf  = lane >> 4;  // 0/1: lane half selects K-subgroup
  const int lmod   = lane & 15;

  const int nBase = blockIdx.x * 128;
  const int mBase = blockIdx.y * 128;

  const v8f vzero = {0.f, 0.f, 0.f, 0.f, 0.f, 0.f, 0.f, 0.f};
  v8f acc[2][4];
  #pragma unroll
  for (int mt = 0; mt < 2; ++mt)
    #pragma unroll
    for (int nt = 0; nt < 4; ++nt) acc[mt][nt] = vzero;

  union Frag { u32x4 q[2]; v16bf v; };

  const int NS = K_DIM / 32;  // 24 K-steps

  // ---- Pipeline prologue ----
  async_copy_x_tile(x, &Xraw[0][0], mBase, 0, tid);       // X raw k=0
  if (wid == 0) {
    tdm_load_w_tile(whi, &Bhi[0][0], nBase, 0);           // W k=0 via TDM
    tdm_load_w_tile(wlo, &Blo[0][0], nBase, 0);
  }
  asm volatile("s_wait_asynccnt 0" ::: "memory");
  if (wid == 0) __builtin_amdgcn_s_wait_tensorcnt(0);
  __syncthreads();

  convert_x_tile(&Xraw[0][0], &Ahi[0][0], &Alo[0][0], tid);  // A k=0 ready
  async_copy_x_tile(x, &Xraw[1][0], mBase, 32, tid);         // X raw k=1
  asm volatile("s_wait_asynccnt 0" ::: "memory");
  __syncthreads();

  int p = 0;
  for (int i = 0; i < NS; ++i) {
    const int k0 = i * 32;

    // Issue next tiles' data movement before computing this one.
    if (i + 1 < NS && wid == 0) {
      tdm_load_w_tile(whi, &Bhi[1 - p][0], nBase, k0 + 32);
      tdm_load_w_tile(wlo, &Blo[1 - p][0], nBase, k0 + 32);
    }
    if (i + 2 < NS)
      async_copy_x_tile(x, &Xraw[i & 1][0], mBase, k0 + 64, tid);

    // ---- Load fragments (ISA 16-bit A 16x32 / B 32x16 layouts) ----
    Frag ah[2], al[2], bh[4], bl[4];
    #pragma unroll
    for (int mt = 0; mt < 2; ++mt) {
      const int ar = wave_m * 32 + mt * 16 + lmod;
      ah[mt].q[0] = *(const u32x4*)&Ahi[p][ar * 32 + lhalf * 8];
      ah[mt].q[1] = *(const u32x4*)&Ahi[p][ar * 32 + 16 + lhalf * 8];
      al[mt].q[0] = *(const u32x4*)&Alo[p][ar * 32 + lhalf * 8];
      al[mt].q[1] = *(const u32x4*)&Alo[p][ar * 32 + 16 + lhalf * 8];
    }
    #pragma unroll
    for (int nt = 0; nt < 4; ++nt) {
      const int br = wave_n * 64 + nt * 16 + lmod;
      bh[nt].q[0] = *(const u32x4*)&Bhi[p][br * 32 + lhalf * 8];
      bh[nt].q[1] = *(const u32x4*)&Bhi[p][br * 32 + 16 + lhalf * 8];
      bl[nt].q[0] = *(const u32x4*)&Blo[p][br * 32 + lhalf * 8];
      bl[nt].q[1] = *(const u32x4*)&Blo[p][br * 32 + 16 + lhalf * 8];
    }

    // ---- 2x4 tiles x 3 split products = 24 v_wmma_f32_16x16x32_bf16 ----
    #pragma unroll
    for (int mt = 0; mt < 2; ++mt) {
      #pragma unroll
      for (int nt = 0; nt < 4; ++nt) {
        acc[mt][nt] = __builtin_amdgcn_wmma_f32_16x16x32_bf16(
            false, ah[mt].v, false, bh[nt].v, (short)0, acc[mt][nt], false, false);
        acc[mt][nt] = __builtin_amdgcn_wmma_f32_16x16x32_bf16(
            false, ah[mt].v, false, bl[nt].v, (short)0, acc[mt][nt], false, false);
        acc[mt][nt] = __builtin_amdgcn_wmma_f32_16x16x32_bf16(
            false, al[mt].v, false, bh[nt].v, (short)0, acc[mt][nt], false, false);
      }
    }

    // Convert next X tile behind the WMMAs (raw arrived before last barrier).
    if (i + 1 < NS)
      convert_x_tile(&Xraw[(i + 1) & 1][0], &Ahi[1 - p][0], &Alo[1 - p][0], tid);

    asm volatile("s_wait_asynccnt 0" ::: "memory");
    if (wid == 0) __builtin_amdgcn_s_wait_tensorcnt(0);
    __syncthreads();
    p ^= 1;
  }

  // ---- Epilogue: fused bias add, M-bounds predicated non-temporal stores ----
  // Output is a 116MB stream; NT stores keep X/W resident in the 192MB L2.
  #pragma unroll
  for (int nt = 0; nt < 4; ++nt) {
    const int col = nBase + wave_n * 64 + nt * 16 + lmod;
    const float bv = bias[col];
    #pragma unroll
    for (int mt = 0; mt < 2; ++mt) {
      #pragma unroll
      for (int r = 0; r < 8; ++r) {
        const int row = mBase + wave_m * 32 + mt * 16 + lhalf * 8 + r;
        if (row < M_ROWS)
          __builtin_nontemporal_store(acc[mt][nt][r] + bv,
                                      out + (size_t)row * N_OUT + col);
      }
    }
  }
}

extern "C" void kernel_launch(void* const* d_in, const int* in_sizes, int n_in,
                              void* d_out, int out_size, void* d_ws, size_t ws_size,
                              hipStream_t stream) {
  const float* x    = (const float*)d_in[0];
  const float* wqkv = (const float*)d_in[1];
  const float* bqkv = (const float*)d_in[2];
  const float* aq   = (const float*)d_in[3];
  const float* bq   = (const float*)d_in[4];
  const float* ak   = (const float*)d_in[5];
  const float* bk   = (const float*)d_in[6];
  const float* av   = (const float*)d_in[7];
  const float* bv   = (const float*)d_in[8];
  float* out = (float*)d_out;

  // Workspace: bf16 hi plane then lo plane of folded weight (7.1 MB total).
  u16* whi = (u16*)d_ws;
  u16* wlo = whi + (size_t)N_OUT * K_DIM;

  lora_fold_split<<<dim3(K_DIM / 256, N_OUT), 256, 0, stream>>>(
      wqkv, aq, bq, ak, bk, av, bv, whi, wlo);

  qkv_lora_gemm<<<dim3(N_OUT / 128, (M_ROWS + 127) / 128), 256, 0, stream>>>(
      x, whi, wlo, bqkv, out);
}
